// MultiHeadAttention_37357625540896
// MI455X (gfx1250) — matrix operand order repaired and checked by reading, not compiled, not run on hardware
//
#include <hip/hip_runtime.h>
#include <hip/hip_bf16.h>

typedef __bf16 bf16_t;
typedef __attribute__((ext_vector_type(16))) __bf16 v16bf;
typedef __attribute__((ext_vector_type(8)))  __bf16 v8bf;
typedef __attribute__((ext_vector_type(8)))  float  v8f;

#define D_MODEL 1024
#define NHEAD   16
#define DKH     64
#define SEQ     2048
#define BATCH   2
#define MROWS   (BATCH * SEQ)   // 4096

#define WMMA_BF16(a, b, c) \
  __builtin_amdgcn_wmma_f32_16x16x32_bf16(false, (a), false, (b), (short)0, (c), false, false)

// ---- fragment loaders (wave32 16x16x32 bf16 layouts) ----
// A fragment: lane holds row m = lane&15; 16 contiguous halves at k = 16*hi .. 16*hi+15
static __device__ __forceinline__ v16bf frag_a(const bf16_t* p, int hi) {
  union { v16bf v; v8bf h[2]; } f;
  f.h[0] = *(const v8bf*)(p + 16 * hi);
  f.h[1] = *(const v8bf*)(p + 16 * hi + 8);
  return f.v;
}
// B fragment: lane holds column n = lane&15 (a row of the transposed operand);
// 16 contiguous halves at k = 16*hi .. 16*hi+15
static __device__ __forceinline__ v16bf frag_b(const bf16_t* p, int hi) {
  return *(const v16bf*)(p + 16 * hi);
}

// ---------------- conversion kernels ----------------
__global__ void k_cvt_bf16(const float* __restrict__ in, bf16_t* __restrict__ out, int n) {
  int i = blockIdx.x * blockDim.x + threadIdx.x;
  if (i < n) out[i] = (bf16_t)in[i];
}

// W[K=1024, N=1024] f32  ->  Wt[N, K] bf16
__global__ void k_transpose_cvt(const float* __restrict__ W, bf16_t* __restrict__ Wt) {
  int i = blockIdx.x * blockDim.x + threadIdx.x;  // 1M threads
  int n = i & (D_MODEL - 1);
  int k = i >> 10;
  Wt[(size_t)n * D_MODEL + k] = (bf16_t)W[(size_t)k * D_MODEL + n];
}

// ---------------- generic bf16 GEMM:  out = A[M,K] * B[K,N] + bias ----------------
// B is supplied transposed: Bt[N,K].  mode: 0 -> bf16 out [B,H,T,dk]
//                                           1 -> bf16 out [B,H,dk,T]  (V transposed)
//                                           2 -> f32  out [M,N]
__global__ void k_gemm(const bf16_t* __restrict__ A, const bf16_t* __restrict__ Bt,
                       const float* __restrict__ bias, void* __restrict__ out,
                       int M, int N, int K, int mode) {
  const int tid  = threadIdx.x;
  const int wid  = tid >> 5;
  const int lane = tid & 31;
  const int lh   = lane & 15;
  const int hi   = lane >> 4;
  const int m0   = blockIdx.y * 128 + (wid >> 1) * 32;  // 4 wave-rows of 32
  const int n0   = blockIdx.x * 128 + (wid & 1) * 64;   // 2 wave-cols of 64

  v8f acc[2][4] = {};

  const bf16_t* arow0 = A + (size_t)(m0 + lh) * K;
  const bf16_t* arow1 = A + (size_t)(m0 + 16 + lh) * K;
  const bf16_t* brow0 = Bt + (size_t)(n0 + 0 * 16 + lh) * K;
  const bf16_t* brow1 = Bt + (size_t)(n0 + 1 * 16 + lh) * K;
  const bf16_t* brow2 = Bt + (size_t)(n0 + 2 * 16 + lh) * K;
  const bf16_t* brow3 = Bt + (size_t)(n0 + 3 * 16 + lh) * K;

  for (int k = 0; k < K; k += 32) {
    v16bf a0 = frag_a(arow0 + k, hi);
    v16bf a1 = frag_a(arow1 + k, hi);
    v16bf b0 = frag_b(brow0 + k, hi);
    v16bf b1 = frag_b(brow1 + k, hi);
    v16bf b2 = frag_b(brow2 + k, hi);
    v16bf b3 = frag_b(brow3 + k, hi);
    acc[0][0] = WMMA_BF16(a0, b0, acc[0][0]);
    acc[1][0] = WMMA_BF16(a1, b0, acc[1][0]);
    acc[0][1] = WMMA_BF16(a0, b1, acc[0][1]);
    acc[1][1] = WMMA_BF16(a1, b1, acc[1][1]);
    acc[0][2] = WMMA_BF16(a0, b2, acc[0][2]);
    acc[1][2] = WMMA_BF16(a1, b2, acc[1][2]);
    acc[0][3] = WMMA_BF16(a0, b3, acc[0][3]);
    acc[1][3] = WMMA_BF16(a1, b3, acc[1][3]);
  }

#pragma unroll
  for (int i = 0; i < 2; i++) {
#pragma unroll
    for (int j = 0; j < 4; j++) {
      const int n = n0 + j * 16 + lh;
      const float bv = bias[n];
#pragma unroll
      for (int v = 0; v < 8; v++) {
        const int m = m0 + i * 16 + v + 8 * hi;
        const float val = acc[i][j][v] + bv;
        if (mode == 2) {
          ((float*)out)[(size_t)m * N + n] = val;
        } else {
          const int b_ = m >> 11;          // / SEQ
          const int t  = m & (SEQ - 1);
          const int h_ = n >> 6;           // / DKH
          const int d  = n & (DKH - 1);
          bf16_t* o = (bf16_t*)out;
          if (mode == 0)
            o[((size_t)(b_ * NHEAD + h_) * SEQ + t) * DKH + d] = (bf16_t)val;
          else
            o[((size_t)(b_ * NHEAD + h_) * DKH + d) * SEQ + t] = (bf16_t)val;
        }
      }
    }
  }
}

// ---------------- flash-style causal attention ----------------
// Q,K : [B,H,T,dk] bf16.  Vt : [B,H,dk,T] bf16.  ctx out: [B*T, C] bf16.
__global__ void k_attn(const bf16_t* __restrict__ Q, const bf16_t* __restrict__ Km,
                       const bf16_t* __restrict__ Vt, bf16_t* __restrict__ ctx) {
  __shared__ __align__(16) bf16_t Plds[8][16][40];  // per-wave 16x32 P tile (padded rows)

  const int tid  = threadIdx.x;
  const int wid  = tid >> 5;
  const int lane = tid & 31;
  const int lh   = lane & 15;
  const int hi   = lane >> 4;
  const int bh   = blockIdx.y;            // 0..31
  const int b_   = bh >> 4;               // / NHEAD
  const int h_   = bh & (NHEAD - 1);
  const int qb   = blockIdx.x * 128 + wid * 16;  // wave-uniform query base

  const bf16_t* Qh = Q  + (size_t)bh * SEQ * DKH;
  const bf16_t* Kh = Km + (size_t)bh * SEQ * DKH;
  const bf16_t* Vh = Vt + (size_t)bh * DKH * SEQ;

  // Q fragments (rows qb..qb+15, dk split into two 32-wide k-steps)
  const v16bf qa0 = frag_a(Qh + (size_t)(qb + lh) * DKH + 0, hi);
  const v16bf qa1 = frag_a(Qh + (size_t)(qb + lh) * DKH + 32, hi);

  v8f o[4] = {};
  float run_max[8], run_sum[8];
#pragma unroll
  for (int v = 0; v < 8; v++) { run_max[v] = -1e30f; run_sum[v] = 0.0f; }

  const float scale = 0.125f;  // 1/sqrt(64)

  for (int kb = 0; kb <= qb + 15; kb += 32) {  // causal: wave-uniform bound
    // ---- scores for keys kb..kb+31 ----
    v16bf kf00 = frag_b(Kh + (size_t)(kb + lh) * DKH + 0, hi);
    v16bf kf01 = frag_b(Kh + (size_t)(kb + lh) * DKH + 32, hi);
    v16bf kf10 = frag_b(Kh + (size_t)(kb + 16 + lh) * DKH + 0, hi);
    v16bf kf11 = frag_b(Kh + (size_t)(kb + 16 + lh) * DKH + 32, hi);
    v8f s0 = {}, s1 = {};
    s0 = WMMA_BF16(qa0, kf00, s0);
    s0 = WMMA_BF16(qa1, kf01, s0);
    s1 = WMMA_BF16(qa0, kf10, s1);
    s1 = WMMA_BF16(qa1, kf11, s1);

    // ---- causal mask + online softmax ----
    float p0[8], p1[8], mloc[8];
#pragma unroll
    for (int v = 0; v < 8; v++) {
      const int qrow = qb + v + 8 * hi;
      const float a = (kb + lh      <= qrow) ? s0[v] * scale : -1e30f;
      const float c = (kb + 16 + lh <= qrow) ? s1[v] * scale : -1e30f;
      p0[v] = a; p1[v] = c;
      mloc[v] = fmaxf(a, c);
    }
#pragma unroll
    for (int off = 1; off < 16; off <<= 1)   // reduce over 16 lanes; halves stay separate
#pragma unroll
      for (int v = 0; v < 8; v++)
        mloc[v] = fmaxf(mloc[v], __shfl_xor(mloc[v], off, 32));

    float alpha[8], rs[8];
#pragma unroll
    for (int v = 0; v < 8; v++) {
      const float nm = fmaxf(run_max[v], mloc[v]);
      alpha[v] = __expf(run_max[v] - nm);
      run_max[v] = nm;
      p0[v] = __expf(p0[v] - nm);
      p1[v] = __expf(p1[v] - nm);
      rs[v] = p0[v] + p1[v];
    }
#pragma unroll
    for (int off = 1; off < 16; off <<= 1)
#pragma unroll
      for (int v = 0; v < 8; v++)
        rs[v] += __shfl_xor(rs[v], off, 32);
#pragma unroll
    for (int v = 0; v < 8; v++) run_sum[v] = run_sum[v] * alpha[v] + rs[v];

    // ---- transpose P (D-layout -> A-layout) via per-wave LDS tile ----
#pragma unroll
    for (int v = 0; v < 8; v++) {
      Plds[wid][v + 8 * hi][lh]      = (bf16_t)p0[v];
      Plds[wid][v + 8 * hi][16 + lh] = (bf16_t)p1[v];
    }
    asm volatile("s_wait_dscnt 0" ::: "memory");
    const v16bf pa = frag_a((const bf16_t*)&Plds[wid][lh][0], hi);

    // ---- O = O*alpha + P @ V  (V columns are contiguous rows of Vt) ----
#pragma unroll
    for (int j = 0; j < 4; j++) {
      v16bf vf = frag_b(Vh + (size_t)(j * 16 + lh) * SEQ + kb, hi);
#pragma unroll
      for (int v = 0; v < 8; v++) o[j][v] *= alpha[v];
      o[j] = WMMA_BF16(pa, vf, o[j]);
    }
  }

  // ---- normalize and write context [B*T, C] ----
#pragma unroll
  for (int v = 0; v < 8; v++) run_sum[v] = 1.0f / run_sum[v];
#pragma unroll
  for (int j = 0; j < 4; j++) {
#pragma unroll
    for (int v = 0; v < 8; v++) {
      const int qrow = qb + v + 8 * hi;
      const int col  = h_ * DKH + j * 16 + lh;
      ctx[(size_t)(b_ * SEQ + qrow) * D_MODEL + col] = (bf16_t)(o[j][v] * run_sum[v]);
    }
  }
}

extern "C" void kernel_launch(void* const* d_in, const int* in_sizes, int n_in,
                              void* d_out, int out_size, void* d_ws, size_t ws_size,
                              hipStream_t stream) {
  (void)in_sizes; (void)n_in; (void)out_size; (void)ws_size;
  const float* x  = (const float*)d_in[0];
  const float* Wq = (const float*)d_in[1];
  const float* bq = (const float*)d_in[2];
  const float* Wk = (const float*)d_in[3];
  const float* bk = (const float*)d_in[4];
  const float* Wv = (const float*)d_in[5];
  const float* bv = (const float*)d_in[6];
  const float* Wo = (const float*)d_in[7];
  const float* bo = (const float*)d_in[8];

  const size_t MB = (size_t)1 << 20;
  char* ws = (char*)d_ws;
  bf16_t* x_bf = (bf16_t*)(ws + 0 * MB);    // 8 MB : [4096,1024]
  bf16_t* Wqt  = (bf16_t*)(ws + 8 * MB);    // 2 MB : [1024,1024] transposed
  bf16_t* Wkt  = (bf16_t*)(ws + 10 * MB);
  bf16_t* Wvt  = (bf16_t*)(ws + 12 * MB);
  bf16_t* Wot  = (bf16_t*)(ws + 14 * MB);
  bf16_t* Qb   = (bf16_t*)(ws + 16 * MB);   // 8 MB : [B,H,T,dk]
  bf16_t* Kb   = (bf16_t*)(ws + 24 * MB);   // 8 MB : [B,H,T,dk]
  bf16_t* Vtb  = (bf16_t*)(ws + 32 * MB);   // 8 MB : [B,H,dk,T]
  bf16_t* ctx  = (bf16_t*)(ws + 40 * MB);   // 8 MB : [B*T, C]

  const int nX = MROWS * D_MODEL;           // 4 Mi elements
  k_cvt_bf16<<<nX / 256, 256, 0, stream>>>(x, x_bf, nX);

  const int nW = D_MODEL * D_MODEL;         // 1 Mi elements
  k_transpose_cvt<<<nW / 256, 256, 0, stream>>>(Wq, Wqt);
  k_transpose_cvt<<<nW / 256, 256, 0, stream>>>(Wk, Wkt);
  k_transpose_cvt<<<nW / 256, 256, 0, stream>>>(Wv, Wvt);
  k_transpose_cvt<<<nW / 256, 256, 0, stream>>>(Wo, Wot);

  dim3 ggrid(D_MODEL / 128, MROWS / 128);   // (8, 32)
  k_gemm<<<ggrid, 256, 0, stream>>>(x_bf, Wqt, bq, Qb,  MROWS, D_MODEL, D_MODEL, 0);
  k_gemm<<<ggrid, 256, 0, stream>>>(x_bf, Wkt, bk, Kb,  MROWS, D_MODEL, D_MODEL, 0);
  k_gemm<<<ggrid, 256, 0, stream>>>(x_bf, Wvt, bv, Vtb, MROWS, D_MODEL, D_MODEL, 1);

  dim3 agrid(SEQ / 128, BATCH * NHEAD);     // (16, 32)
  k_attn<<<agrid, 256, 0, stream>>>(Qb, Kb, Vtb, ctx);

  k_gemm<<<ggrid, 256, 0, stream>>>(ctx, Wot, bo, d_out, MROWS, D_MODEL, D_MODEL, 2);
}
